// AttentionV1_82119774699927
// MI455X (gfx1250) — matrix operand order repaired and checked by reading, not compiled, not run on hardware
//
#include <hip/hip_runtime.h>
#include <hip/hip_bf16.h>

// ---------------------------------------------------------------------------
// AttentionV1 for MI455X (gfx1250, wave32, WMMA + async LDS copies).
//   z = Wz @ ( V @ softmax_q( K^T Q ) ) + bz + x
// Streaming two-pass per-key softmax (never materializes the 268MB score
// matrix). All GEMMs: v_wmma_f32_16x16x32_bf16 (bf16 in, f32 accum).
// Q/K are produced pre-transposed ([n][ch], ch contiguous) so every LDS tile
// is staged with global_load_async_to_lds_b128 (ASYNCcnt). Softmax stats use
// a per-lane online update (no shuffles in the sweep); the final cross-lane
// merge is butterfly-max, one rescale exp, butterfly-add.
// ---------------------------------------------------------------------------

#define B_  4
#define C_  256
#define CH_ 128
#define N_  4096

typedef unsigned short u16;
typedef unsigned int   u32;
typedef __attribute__((ext_vector_type(16))) __bf16 v16bf;
typedef __attribute__((ext_vector_type(8)))  float  v8f;
typedef __attribute__((ext_vector_type(4)))  u32    u32x4;
typedef __attribute__((ext_vector_type(2)))  u32    u32x2;

union Frag { v16bf v; u32 u[8]; u32x4 q[2]; };

__device__ inline u16 f2bf(float f) {
    u32 u = __float_as_uint(f);
    u32 r = (u + 0x7FFFu + ((u >> 16) & 1u)) >> 16;   // round-to-nearest-even
    return (u16)r;
}
__device__ inline u32 pack2bf(float lo, float hi) {
    return (u32)f2bf(lo) | ((u32)f2bf(hi) << 16);
}
__device__ inline v8f zero8() {
    v8f v;
#pragma unroll
    for (int i = 0; i < 8; ++i) v[i] = 0.0f;
    return v;
}
__device__ inline v8f wmma_bf16(v16bf a, v16bf b, v8f c) {
    // D = A(16x32 bf16) * B(32x16 bf16) + C(16x16 f32)
    return __builtin_amdgcn_wmma_f32_16x16x32_bf16(false, a, false, b,
                                                   (short)0, c, false, false);
}

// --- CDNA5 async global->LDS copy (16B per lane), tracked by ASYNCcnt -------
__device__ inline u32 lds_addr(const void* p) {
    // generic -> LDS address space; ptrtoint yields the in-group-segment offset
    return (u32)(size_t)(const __attribute__((address_space(3))) char*)p;
}
__device__ inline void async_b128(void* lds, const void* gsrc) {
    asm volatile("global_load_async_to_lds_b128 %0, %1, off"
                 :: "v"(lds_addr(lds)), "v"(gsrc) : "memory");
}
__device__ inline void wait_async() {
#if defined(__has_builtin)
#if __has_builtin(__builtin_amdgcn_s_wait_asynccnt)
    __builtin_amdgcn_s_wait_asynccnt(0);
    return;
#endif
#endif
    asm volatile("s_wait_asynccnt 0x0" ::: "memory");
}

// A-fragment (16x32, bf16) from a [row][contraction] tile (stride u16, even).
__device__ inline v16bf load_a(const u16* base, int row0, int stride, int k0) {
    const int lane = threadIdx.x & 31;
    const int half = lane >> 4;
    const int m    = lane & 15;
    const u32* p = (const u32*)(base + (size_t)(row0 + m) * stride);
    const int kb = k0 + 8 * half;
    Frag f;
#pragma unroll
    for (int j = 0; j < 4; ++j) f.u[j]     = p[(kb + 2 * j) >> 1];
#pragma unroll
    for (int j = 0; j < 4; ++j) f.u[4 + j] = p[(kb + 16 + 2 * j) >> 1];
    return f.v;
}

// B-fragment (32x16, bf16) from a [col][contraction] tile.
__device__ inline v16bf load_b(const u16* base, int col0, int stride, int k0) {
    const int lane = threadIdx.x & 31;
    const int half = lane >> 4;
    const int n    = lane & 15;
    const u16* p = base + (size_t)(col0 + n) * stride + k0 + 8 * half;
    Frag f;
    f.q[0] = *(const u32x4*)p;
    f.q[1] = *(const u32x4*)(p + 16);
    return f.v;
}

// ---------------------------------------------------------------------------
// f32 [CH*C] weights -> bf16, vectorized (float4 -> 4x bf16 per thread)
__global__ void cvt_bf16_kernel(const float* __restrict__ src,
                                u16* __restrict__ dst, int n) {
    int i = (blockIdx.x * blockDim.x + threadIdx.x) * 4;
    if (i < n) {
        float4 v = *(const float4*)(src + i);
        u32x2 o;
        o[0] = pack2bf(v.x, v.y);
        o[1] = pack2bf(v.z, v.w);
        *(u32x2*)(dst + i) = o;
    }
}

// f32 x[b][c][n] -> bf16 xT[b][n][c] via 4x4 register-blocked transpose.
__global__ __launch_bounds__(256) void cvt_tr_kernel(
    const float* __restrict__ src, u16* __restrict__ dstT) {
    const int n0 = blockIdx.x * 64;
    const int c0 = blockIdx.y * 64;
    const int b  = blockIdx.z;
    const size_t bIn  = (size_t)b * C_ * N_;
    const size_t bOut = (size_t)b * N_ * C_;
    const int t  = threadIdx.x;
    const int bn = (t & 15) * 4;   // 4 n-values (contiguous in src)
    const int bc = (t >> 4) * 4;   // 4 c-rows

    const float* s = src + bIn + (size_t)(c0 + bc) * N_ + n0 + bn;
    float4 r0 = *(const float4*)(s);
    float4 r1 = *(const float4*)(s + N_);
    float4 r2 = *(const float4*)(s + 2 * N_);
    float4 r3 = *(const float4*)(s + 3 * N_);

    u16* d = dstT + bOut + (size_t)(n0 + bn) * C_ + c0 + bc;
    u32x2 o;
    o[0] = pack2bf(r0.x, r1.x); o[1] = pack2bf(r2.x, r3.x);
    *(u32x2*)(d) = o;
    o[0] = pack2bf(r0.y, r1.y); o[1] = pack2bf(r2.y, r3.y);
    *(u32x2*)(d + C_) = o;
    o[0] = pack2bf(r0.z, r1.z); o[1] = pack2bf(r2.z, r3.z);
    *(u32x2*)(d + 2 * C_) = o;
    o[0] = pack2bf(r0.w, r1.w); o[1] = pack2bf(r2.w, r3.w);
    *(u32x2*)(d + 3 * C_) = o;
}

// ---------------------------------------------------------------------------
// Projection GEMM: out[o,n] = sum_c W[o,c] * X[n,c] + bias[o]
// TR=true  -> store OutT[n][o] (one b128 per accumulator; used for Q, K)
// TR=false -> store Out[o][n]  (natural; used for V)
template <bool TR>
__global__ __launch_bounds__(256) void proj_kernel(
    const u16* __restrict__ XbfT, const u16* __restrict__ Wbf,
    const float* __restrict__ bias, u16* __restrict__ Out) {
    __shared__ __align__(16) u16 XT[64 * 264];  // [n][c], full C staged once

    const int n0 = blockIdx.x * 64;
    const int o0 = blockIdx.y * 64;
    const int b  = blockIdx.z;
    const int t = threadIdx.x, w = t >> 5, lane = t & 31;
    const int half = lane >> 4, ln16 = lane & 15;
    const size_t bX = (size_t)b * N_ * C_;

#pragma unroll
    for (int i = 0; i < 8; ++i) {               // 2048 async b128 copies total
        int idx = i * 256 + t;
        int n = idx >> 5, c8 = (idx & 31) * 8;
        async_b128(XT + n * 264 + c8, XbfT + bX + (size_t)(n0 + n) * C_ + c8);
    }
    wait_async();
    __syncthreads();

    const int osub = o0 + (w & 3) * 16;
    const int nh   = (w >> 2) * 32;

    v8f acc0 = zero8(), acc1 = zero8();
    for (int c0 = 0; c0 < C_; c0 += 32) {
        v16bf a  = load_a(Wbf, osub, C_, c0);   // W rows direct from global
        v16bf b0 = load_b(XT, nh, 264, c0);
        v16bf b1 = load_b(XT, nh + 16, 264, c0);
        acc0 = wmma_bf16(a, b0, acc0);
        acc1 = wmma_bf16(a, b1, acc1);
    }

    if constexpr (TR) {                         // OutT[n][o], o contiguous
        const size_t bO = (size_t)b * N_ * CH_;
        Frag f0, f1;
#pragma unroll
        for (int r = 0; r < 8; r += 2) {
            int orow = osub + 8 * half + r;
            f0.u[r >> 1] = pack2bf(acc0[r] + bias[orow], acc0[r + 1] + bias[orow + 1]);
            f1.u[r >> 1] = pack2bf(acc1[r] + bias[orow], acc1[r + 1] + bias[orow + 1]);
        }
        *(u32x4*)(Out + bO + (size_t)(n0 + nh + ln16) * CH_ + osub + 8 * half)      = f0.q[0];
        *(u32x4*)(Out + bO + (size_t)(n0 + nh + 16 + ln16) * CH_ + osub + 8 * half) = f1.q[0];
    } else {                                    // Out[o][n]
        const size_t bO = (size_t)b * CH_ * N_;
#pragma unroll
        for (int r = 0; r < 8; ++r) {
            int orow = osub + r + 8 * half;
            float bv = bias[orow];
            Out[bO + (size_t)orow * N_ + n0 + nh + ln16]      = f2bf(acc0[r] + bv);
            Out[bO + (size_t)orow * N_ + n0 + nh + 16 + ln16] = f2bf(acc1[r] + bv);
        }
    }
}

// ---------------------------------------------------------------------------
// Pass 1: per-key online softmax stats over the query axis.
//   S[k,q] = sum_c KT[k,c]*QT[q,c];  m[k]=max_q S;  l[k]=sum_q exp(S-m)
// Each lane keeps private (m,l) over its own q-columns; final merge is
// butterfly-max, one rescale exp per row, butterfly-add.
__global__ __launch_bounds__(256) void attn_stats_kernel(
    const u16* __restrict__ KbfT, const u16* __restrict__ QbfT,
    float* __restrict__ Mbuf, float* __restrict__ Lbuf) {
    __shared__ __align__(16) u16 KT[64 * 136];  // [k][c], staged once
    __shared__ __align__(16) u16 QT[64 * 136];  // [q][c], per q-chunk
    __shared__ float Mred[2][64];
    __shared__ float Lred[2][64];

    const int k0g = blockIdx.x * 64;
    const int b   = blockIdx.y;
    const int t = threadIdx.x, w = t >> 5, lane = t & 31;
    const int half = lane >> 4, ln16 = lane & 15;
    const size_t bNT = (size_t)b * N_ * CH_;

#pragma unroll
    for (int i = 0; i < 4; ++i) {               // K tile: rows k, c contiguous
        int idx = i * 256 + t;
        int k = idx >> 4, c8 = (idx & 15) * 8;
        async_b128(KT + k * 136 + c8, KbfT + bNT + (size_t)(k0g + k) * CH_ + c8);
    }

    const int ksub  = (w & 3) * 16;
    const int qhsel = w >> 2;
    float m_run[8], l_run[8];
#pragma unroll
    for (int r = 0; r < 8; ++r) { m_run[r] = -1e30f; l_run[r] = 0.0f; }

    for (int q0 = 0; q0 < N_; q0 += 64) {
        __syncthreads();
#pragma unroll
        for (int i = 0; i < 4; ++i) {
            int idx = i * 256 + t;
            int q = idx >> 4, c8 = (idx & 15) * 8;
            async_b128(QT + q * 136 + c8, QbfT + bNT + (size_t)(q0 + q) * CH_ + c8);
        }
        wait_async();
        __syncthreads();

        v8f s0 = zero8(), s1 = zero8();
        for (int c0 = 0; c0 < CH_; c0 += 32) {
            v16bf a  = load_a(KT, ksub, 136, c0);
            v16bf b0 = load_b(QT, qhsel * 32, 136, c0);
            v16bf b1 = load_b(QT, qhsel * 32 + 16, 136, c0);
            s0 = wmma_bf16(a, b0, s0);
            s1 = wmma_bf16(a, b1, s1);
        }
#pragma unroll
        for (int r = 0; r < 8; ++r) {           // per-lane online update
            float mn = fmaxf(m_run[r], fmaxf(s0[r], s1[r]));
            l_run[r] = l_run[r] * __expf(m_run[r] - mn)
                     + __expf(s0[r] - mn) + __expf(s1[r] - mn);
            m_run[r] = mn;
        }
    }

    // merge across the 16 q-lanes: butterfly max, one rescale, butterfly add
#pragma unroll
    for (int r = 0; r < 8; ++r) {
        float M = m_run[r];
#pragma unroll
        for (int mask = 1; mask <= 8; mask <<= 1)
            M = fmaxf(M, __shfl_xor(M, mask, 32));
        float l = l_run[r] * __expf(m_run[r] - M);
#pragma unroll
        for (int mask = 1; mask <= 8; mask <<= 1)
            l += __shfl_xor(l, mask, 32);
        m_run[r] = M;
        l_run[r] = l;
    }

    if (ln16 == 0) {                            // lanes 0/16 publish 8 rows each
#pragma unroll
        for (int r = 0; r < 8; ++r) {
            int kloc = ksub + 8 * half + r;
            Mred[qhsel][kloc] = m_run[r];
            Lred[qhsel][kloc] = l_run[r];
        }
    }
    __syncthreads();
    if (t < 64) {                               // combine the two q-half waves
        float m0 = Mred[0][t], m1 = Mred[1][t];
        float mm = fmaxf(m0, m1);
        float ll = Lred[0][t] * __expf(m0 - mm) + Lred[1][t] * __expf(m1 - mm);
        Mbuf[b * N_ + k0g + t] = mm;
        Lbuf[b * N_ + k0g + t] = ll;
    }
}

// ---------------------------------------------------------------------------
// Pass 2 (fused): per q-tile of 64, sweep k chunks:
//   S = K^T Q -> P = exp(S-m[k])/l[k] -> y += V @ P; then z = Wz@y + bz + x.
__global__ __launch_bounds__(256) void attn_main_kernel(
    const u16* __restrict__ QbfT, const u16* __restrict__ KbfT,
    const u16* __restrict__ Vbf, const u16* __restrict__ Wzbf,
    const float* __restrict__ bz, const float* __restrict__ x,
    const float* __restrict__ Mbuf, const float* __restrict__ Lbuf,
    float* __restrict__ out) {
    __shared__ __align__(16) u16 QT[64 * 136];  // [q][c], staged once
    __shared__ __align__(16) u16 KT[64 * 136];  // [k][c] per chunk; reused as YT
    __shared__ __align__(16) u16 VT[128 * 72];  // [c][k] per chunk
    __shared__ __align__(16) u16 PT[64 * 72];   // [q][k] per chunk
    __shared__ float mCh[64];
    __shared__ float rlCh[64];

    const int n0 = blockIdx.x * 64;
    const int b  = blockIdx.y;
    const int t = threadIdx.x, w = t >> 5, lane = t & 31;
    const int half = lane >> 4, ln16 = lane & 15;
    const size_t bNT  = (size_t)b * N_ * CH_;
    const size_t bQKV = (size_t)b * CH_ * N_;

#pragma unroll
    for (int i = 0; i < 4; ++i) {               // Q tile once (async)
        int idx = i * 256 + t;
        int q = idx >> 4, c8 = (idx & 15) * 8;
        async_b128(QT + q * 136 + c8, QbfT + bNT + (size_t)(n0 + q) * CH_ + c8);
    }

    v8f accY[4];
#pragma unroll
    for (int i = 0; i < 4; ++i) accY[i] = zero8();

    const int ksub = (w & 3) * 16;              // S-phase wave mapping
    const int qh   = (w >> 2) * 32;

    for (int kc = 0; kc < N_; kc += 64) {
        __syncthreads();
#pragma unroll
        for (int i = 0; i < 4; ++i) {           // K tile (async)
            int idx = i * 256 + t;
            int k = idx >> 4, c8 = (idx & 15) * 8;
            async_b128(KT + k * 136 + c8, KbfT + bNT + (size_t)(kc + k) * CH_ + c8);
        }
#pragma unroll
        for (int i = 0; i < 4; ++i) {           // V tile (async, natural layout)
            int idx = i * 256 + t;
            int c = idx >> 3, k8 = (idx & 7) * 8;
            async_b128(VT + c * 72 + k8, Vbf + bQKV + (size_t)c * N_ + kc + k8);
        }
        if (t < 64) {
            mCh[t]  = Mbuf[b * N_ + kc + t];
            rlCh[t] = 1.0f / Lbuf[b * N_ + kc + t];
        }
        wait_async();
        __syncthreads();

        // S subtile (16k x 32q per wave)
        v8f s0 = zero8(), s1 = zero8();
        for (int c0 = 0; c0 < CH_; c0 += 32) {
            v16bf a  = load_a(KT, ksub, 136, c0);
            v16bf b0 = load_b(QT, qh, 136, c0);
            v16bf b1 = load_b(QT, qh + 16, 136, c0);
            s0 = wmma_bf16(a, b0, s0);
            s1 = wmma_bf16(a, b1, s1);
        }
        // P = exp(S - m[k]) / l[k] -> PT[q][k] (one b128 store per accumulator)
        {
            Frag f0, f1;
#pragma unroll
            for (int r = 0; r < 8; r += 2) {
                int k0i = ksub + 8 * half + r;
                f0.u[r >> 1] = pack2bf(__expf(s0[r] - mCh[k0i]) * rlCh[k0i],
                                       __expf(s0[r + 1] - mCh[k0i + 1]) * rlCh[k0i + 1]);
                f1.u[r >> 1] = pack2bf(__expf(s1[r] - mCh[k0i]) * rlCh[k0i],
                                       __expf(s1[r + 1] - mCh[k0i + 1]) * rlCh[k0i + 1]);
            }
            int q0l = qh + ln16;
            *(u32x4*)(PT + (size_t)q0l * 72 + ksub + 8 * half)        = f0.q[0];
            *(u32x4*)(PT + (size_t)(q0l + 16) * 72 + ksub + 8 * half) = f1.q[0];
        }
        __syncthreads();

        // y += V @ P  (wave w owns c-rows w*16..+16, all 4 q subtiles)
        for (int kk0 = 0; kk0 < 64; kk0 += 32) {
            v16bf a = load_a(VT, w * 16, 72, kk0);
#pragma unroll
            for (int qs = 0; qs < 4; ++qs) {
                v16bf bb = load_b(PT, qs * 16, 72, kk0);
                accY[qs] = wmma_bf16(a, bb, accY[qs]);
            }
        }
    }

    // Epilogue: y -> bf16 YT[q][c] (reuse KT), then z = Wz@y + bz + x
    __syncthreads();
    u16* YT = KT;
#pragma unroll
    for (int qs = 0; qs < 4; ++qs) {
        Frag f;
#pragma unroll
        for (int r = 0; r < 8; r += 2)
            f.u[r >> 1] = pack2bf(accY[qs][r], accY[qs][r + 1]);
        int q = qs * 16 + ln16;
        *(u32x4*)(YT + (size_t)q * 136 + w * 16 + 8 * half) = f.q[0];
    }
    __syncthreads();

    v8f accZ[2][4];
#pragma unroll
    for (int s = 0; s < 2; ++s)
#pragma unroll
        for (int qs = 0; qs < 4; ++qs) accZ[s][qs] = zero8();

    const int cob = w * 32;                     // wave w -> output rows [cob, cob+32)
    for (int c0 = 0; c0 < CH_; c0 += 32) {
        v16bf a0 = load_a(Wzbf, cob, 128, c0);  // Wz rows direct from global
        v16bf a1 = load_a(Wzbf, cob + 16, 128, c0);
#pragma unroll
        for (int qs = 0; qs < 4; ++qs) {
            v16bf bb = load_b(YT, qs * 16, 136, c0);
            accZ[0][qs] = wmma_bf16(a0, bb, accZ[0][qs]);
            accZ[1][qs] = wmma_bf16(a1, bb, accZ[1][qs]);
        }
    }
    const size_t bX = (size_t)b * C_ * N_;
#pragma unroll
    for (int s = 0; s < 2; ++s) {
#pragma unroll
        for (int qs = 0; qs < 4; ++qs) {
#pragma unroll
            for (int r = 0; r < 8; ++r) {
                int co = cob + s * 16 + r + 8 * half;
                int n  = n0 + qs * 16 + ln16;
                size_t o = bX + (size_t)co * N_ + n;
                out[o] = accZ[s][qs][r] + bz[co] + x[o];
            }
        }
    }
}

// ---------------------------------------------------------------------------
extern "C" void kernel_launch(void* const* d_in, const int* in_sizes, int n_in,
                              void* d_out, int out_size, void* d_ws, size_t ws_size,
                              hipStream_t stream) {
    (void)in_sizes; (void)n_in; (void)out_size; (void)ws_size;
    const float* x    = (const float*)d_in[0];
    const float* aux  = (const float*)d_in[1];
    const float* wq_w = (const float*)d_in[2];
    const float* wq_b = (const float*)d_in[3];
    const float* wk_w = (const float*)d_in[4];
    const float* wk_b = (const float*)d_in[5];
    const float* wv_w = (const float*)d_in[6];
    const float* wv_b = (const float*)d_in[7];
    const float* wz_w = (const float*)d_in[8];
    const float* wz_b = (const float*)d_in[9];
    float* out = (float*)d_out;

    const size_t XY  = (size_t)B_ * C_ * N_;   // 4,194,304
    const size_t QS  = (size_t)B_ * CH_ * N_;  // 2,097,152
    const size_t WSZ = (size_t)CH_ * C_;       // 32,768

    u16* XbfT = (u16*)d_ws;          // [B][N][C]   (transposed, bf16)
    u16* AbfT = XbfT + XY;           // [B][N][C]
    u16* QbT  = AbfT + XY;           // [B][N][CH]  (transposed)
    u16* KbT  = QbT + QS;            // [B][N][CH]  (transposed)
    u16* Vb   = KbT + QS;            // [B][CH][N]  (natural)
    u16* Wq   = Vb + QS;
    u16* Wk   = Wq + WSZ;
    u16* Wv   = Wk + WSZ;
    u16* Wz   = Wv + WSZ;
    float* Mb = (float*)(Wz + WSZ);
    float* Lb = Mb + (size_t)B_ * N_;

    dim3 gTr(N_ / 64, C_ / 64, B_);
    cvt_tr_kernel<<<gTr, 256, 0, stream>>>(x, XbfT);
    cvt_tr_kernel<<<gTr, 256, 0, stream>>>(aux, AbfT);
    cvt_bf16_kernel<<<32, 256, 0, stream>>>(wq_w, Wq, (int)WSZ);
    cvt_bf16_kernel<<<32, 256, 0, stream>>>(wk_w, Wk, (int)WSZ);
    cvt_bf16_kernel<<<32, 256, 0, stream>>>(wv_w, Wv, (int)WSZ);
    cvt_bf16_kernel<<<32, 256, 0, stream>>>(wz_w, Wz, (int)WSZ);

    dim3 gProj(N_ / 64, CH_ / 64, B_);
    proj_kernel<true><<<gProj, 256, 0, stream>>>(XbfT, Wq, wq_b, QbT);
    proj_kernel<true><<<gProj, 256, 0, stream>>>(AbfT, Wk, wk_b, KbT);
    proj_kernel<false><<<gProj, 256, 0, stream>>>(XbfT, Wv, wv_b, Vb);

    attn_stats_kernel<<<dim3(N_ / 64, B_), 256, 0, stream>>>(KbT, QbT, Mb, Lb);
    attn_main_kernel<<<dim3(N_ / 64, B_), 256, 0, stream>>>(QbT, KbT, Vb, Wz, wz_b,
                                                            x, Mb, Lb, out);
}
